// NeuralFVSolver_86638080295380
// MI455X (gfx1250) — matrix operand order repaired and checked by reading, not compiled, not run on hardware
//
#include <hip/hip_runtime.h>
#include <hip/hip_bf16.h>

// NeuralFVSolver on MI455X (gfx1250, wave32, WMMA).
// One workgroup per batch row; entire 63-step time loop in-kernel with state in LDS.
// MLP layers run on V_WMMA_F32_16X16X32_F16 (f16 in, f32 accumulate).
// GELU/exp are branch-free (v_rcp_f32 + v_exp_f32 + FMA chain).
// M-tile loops are kept rolled (#pragma unroll 1) to bound register pressure at
// 4 waves/SIMD and avoid scratch spills around the WMMAs.

#define KRAD   3
#define SIGMAC 0.05f
#define DXC    0.02f
#define HID    64
#define NXC    1024
#define NTC    64
#define NBATCH 16
#define INFEAT 22   // (2K+1)*3 + 1

#define NTHREADS 512
#define NWAVES   16          // wave32
#define TILES_PER_WAVE 4     // 64 tiles of 16 cells

typedef __attribute__((ext_vector_type(16))) _Float16 v16h;
typedef __attribute__((ext_vector_type(8)))  _Float16 v8h;
typedef __attribute__((ext_vector_type(8)))  float    v8f;

#define LOG2E 1.4426950408889634f

// Branch-free exact-GELU: 0.5*x*(1+erf(x/sqrt2)) via Abramowitz-Stegun 7.1.26
// (max abs err ~1.5e-7, far below the f16 activation rounding). Computed in the
// cancellation-safe form q = 0.5*poly*exp(-z^2) = Phi(-|x|).
__device__ __forceinline__ float gelu_fast(float x) {
    float z  = fabsf(x) * 0.70710678118654752440f;     // |x|/sqrt(2)
    float t  = __builtin_amdgcn_rcpf(fmaf(0.3275911f, z, 1.0f));
    float p  = t * fmaf(t, fmaf(t, fmaf(t, fmaf(t, 1.061405429f, -1.453152027f),
                                        1.421413741f), -0.284496736f), 0.254829592f);
    float e  = __builtin_amdgcn_exp2f(z * z * -LOG2E); // exp(-z^2), raw v_exp_f32
    float q  = 0.5f * p * e;                           // Phi(-|x|)
    float cdf = (x >= 0.0f) ? (1.0f - q) : q;          // v_cndmask, no branch
    return x * cdf;
}

__device__ __forceinline__ v8f wmma_f16(v16h a, v16h b, v8f c) {
    // D = A(16x32 f16) * B(32x16 f16) + C(16x16 f32)
    return __builtin_amdgcn_wmma_f32_16x16x32_f16(
        /*neg_a=*/false, a, /*neg_b=*/false, b,
        /*c_mod=*/(short)0, c, /*reuse_a=*/false, /*reuse_b=*/false);
}

// A-tile load in contiguous K order (matches B operands): lanes 0-15: row=lane, K=0..15;
// lanes 16-31: row=lane-16, K=16..31. Weights stored [row][K] f16 in LDS.
__device__ __forceinline__ v16h load_a_tile(const _Float16* w, int kdim, int mt, int kt, int lane) {
    const int row  = mt * 16 + (lane & 15);
    const int base = row * kdim + kt * 32 + ((lane & 16) ? 16 : 0);
    v8h lo = *reinterpret_cast<const v8h*>(w + base);        // K seg0 (8 f16, 16B aligned)
    v8h hi = *reinterpret_cast<const v8h*>(w + base + 8);    // K seg1
    return __builtin_shufflevector(lo, hi, 0,1,2,3,4,5,6,7,8,9,10,11,12,13,14,15);
}

__global__ __launch_bounds__(NTHREADS, 1)
void nfv_kernel(const float* __restrict__ gin, const float* __restrict__ dtp,
                const float* __restrict__ W0, const float* __restrict__ b0,
                const float* __restrict__ W1, const float* __restrict__ b1,
                const float* __restrict__ W2, const float* __restrict__ b2,
                const float* __restrict__ W3, const float* __restrict__ b3,
                float* __restrict__ out)
{
    __shared__ __align__(32) float    sState[2][NXC];   // double-buffered state
    __shared__ __align__(32) float    sProx[NXC];
    __shared__ __align__(32) float    sMax[NXC];        // prefix-max scan (shock idx)
    __shared__ __align__(32) float    sMin[NXC];        // suffix-min scan (shock idx)
    __shared__ __align__(32) _Float16 sW0[HID * 32];    // K padded 22 -> 32
    __shared__ __align__(32) _Float16 sW1[HID * HID];
    __shared__ __align__(32) _Float16 sW2[HID * HID];
    __shared__ __align__(32) float    sW3[HID];
    __shared__ __align__(32) float    sB0[HID];
    __shared__ __align__(32) float    sB1[HID];
    __shared__ __align__(32) float    sB2[HID];
    __shared__ __align__(32) _Float16 sAct[NWAVES][16 * 64]; // per-wave activation scratch

    const int b    = blockIdx.x;
    const int tid  = threadIdx.x;
    const int lane = tid & 31;
    const int wave = tid >> 5;
    const bool hi  = (lane & 16) != 0;
    const int l15  = lane & 15;
    const int mrow = hi ? 8 : 0;

    // ---- Stage weights (f32 -> f16, WMMA-friendly [row][K] layout) ----
    for (int i = tid; i < HID * 32; i += NTHREADS) {
        int r = i >> 5, c = i & 31;
        sW0[i] = (_Float16)((c < INFEAT) ? W0[r * INFEAT + c] : 0.0f);
    }
    for (int i = tid; i < HID * HID; i += NTHREADS) {
        sW1[i] = (_Float16)W1[i];
        sW2[i] = (_Float16)W2[i];
    }
    for (int i = tid; i < HID; i += NTHREADS) {
        sW3[i] = W3[i]; sB0[i] = b0[i]; sB1[i] = b1[i]; sB2[i] = b2[i];
    }

    // ---- Initial state (t = 0 slice) + write t=0 output ----
    const size_t bOff = (size_t)b * NTC * NXC;
    for (int i = tid; i < NXC; i += NTHREADS) {
        float v = gin[bOff + i];
        sState[0][i] = v;
        out[bOff + i] = v;
    }
    const float dtb  = dtp[b];
    const float b3v  = b3[0];
    const float dtdx = dtb / DXC;
    __syncthreads();

    int p = 0;
    for (int step = 1; step < NTC; ++step) {
        const float* sc = sState[p];
        float*       sn = sState[p ^ 1];

        // ---- Shock flags: is_shock[m] <=> state[m+1] > state[m] (algebraic reduction) ----
        for (int i = tid; i < NXC; i += NTHREADS) {
            bool shock = (i < NXC - 1) && (sc[i + 1] > sc[i]);
            sMax[i] = shock ? (float)i : -1e9f;
            sMin[i] = shock ? (float)i :  1e9f;
        }
        __syncthreads();

        // ---- Hillis-Steele scans: prefix-max and suffix-min of shock indices ----
        for (int off = 1; off < NXC; off <<= 1) {
            float vmax[2], vmin[2];
            #pragma unroll
            for (int j = 0; j < 2; ++j) {
                int i = tid + j * NTHREADS;
                float m = sMax[i];
                if (i >= off) m = fmaxf(m, sMax[i - off]);
                vmax[j] = m;
                float n = sMin[i];
                if (i + off < NXC) n = fminf(n, sMin[i + off]);
                vmin[j] = n;
            }
            __syncthreads();
            #pragma unroll
            for (int j = 0; j < 2; ++j) {
                sMax[tid + j * NTHREADS] = vmax[j];
                sMin[tid + j * NTHREADS] = vmin[j];
            }
            __syncthreads();
        }
        const float anyShock = (sMax[NXC - 1] > -1e8f) ? 1.0f : 0.0f;

        // ---- Proximity: exp(-nearest_shock_dist / sigma) * any_shock ----
        for (int i = tid; i < NXC; i += NTHREADS) {
            float lm = (i > 0) ? sMax[i - 1] : -1e9f;
            float rm = sMin[i];
            float dl = ((float)i - lm - 0.5f) * DXC;
            float dr = (rm + 0.5f - (float)i) * DXC;
            float md = fminf(dl, dr);
            // exp(-md/sigma) via raw v_exp_f32; huge md underflows cleanly to 0
            sProx[i] = __builtin_amdgcn_exp2f(md * (-LOG2E / SIGMAC)) * anyShock;
        }
        __syncthreads();

        // ---- MLP via WMMA: each wave handles TILES_PER_WAVE tiles of 16 cells ----
        _Float16* myAct = sAct[wave];
        #pragma unroll 1
        for (int ti = 0; ti < TILES_PER_WAVE; ++ti) {
            const int cb = (wave * TILES_PER_WAVE + ti) * 16;
            const int c  = cb + l15;

            // Feature B-operand (32x16): lanes 0-15 hold K0..15, lanes 16-31 hold K16..31.
            // K0-6: stencil state, K7-13: 1-2*state, K14-20: stencil prox, K21: dt, K22-31: 0.
            v16h fB;
            if (!hi) {
                #pragma unroll
                for (int s = 0; s < 7; ++s) {
                    int ci = c + s - KRAD;
                    ci = ci < 0 ? 0 : (ci > NXC - 1 ? NXC - 1 : ci);
                    float sv = sc[ci];
                    fB[s]     = (_Float16)sv;
                    fB[s + 7] = (_Float16)(1.0f - 2.0f * sv);
                }
                int c0 = (c - 3 < 0) ? 0 : c - 3;
                int c1 = (c - 2 < 0) ? 0 : c - 2;
                fB[14] = (_Float16)sProx[c0];
                fB[15] = (_Float16)sProx[c1];
            } else {
                #pragma unroll
                for (int s = 0; s < 5; ++s) {
                    int ci = c + s - 1;
                    ci = ci < 0 ? 0 : (ci > NXC - 1 ? NXC - 1 : ci);
                    fB[s] = (_Float16)sProx[ci];
                }
                fB[5] = (_Float16)dtb;
                #pragma unroll
                for (int s = 6; s < 16; ++s) fB[s] = (_Float16)0.0f;
            }

            // ---- Layer 0: 64x22 (K padded to 32), 4 M-tiles, bias in C ----
            // Rolled: keeps live set small (one A-tile + one accumulator at a time).
            #pragma unroll 1
            for (int t = 0; t < 4; ++t) {
                v16h a  = load_a_tile(sW0, 32, t, 0, lane);
                v8f acc = *reinterpret_cast<const v8f*>(sB0 + t * 16 + mrow);
                acc = wmma_f16(a, fB, acc);
                v8h g;
                #pragma unroll
                for (int r = 0; r < 8; ++r) g[r] = (_Float16)gelu_fast(acc[r]);
                // D layout: vgpr r -> M = t*16 + r (+8 for hi lanes); 8 rows -> one b128 store
                *reinterpret_cast<v8h*>(myAct + l15 * 64 + t * 16 + mrow) = g;
            }

            // ---- Layer 1: 64x64, 2 K-tiles x 4 M-tiles ----
            {
                v16h bk0 = *reinterpret_cast<const v16h*>(myAct + l15 * 64 + (hi ? 16 : 0));
                v16h bk1 = *reinterpret_cast<const v16h*>(myAct + l15 * 64 + 32 + (hi ? 16 : 0));
                #pragma unroll 1
                for (int t = 0; t < 4; ++t) {
                    v16h a0 = load_a_tile(sW1, 64, t, 0, lane);
                    v16h a1 = load_a_tile(sW1, 64, t, 1, lane);
                    v8f acc = *reinterpret_cast<const v8f*>(sB1 + t * 16 + mrow);
                    acc = wmma_f16(a0, bk0, acc);
                    acc = wmma_f16(a1, bk1, acc);
                    v8h g;
                    #pragma unroll
                    for (int r = 0; r < 8; ++r) g[r] = (_Float16)gelu_fast(acc[r]);
                    *reinterpret_cast<v8h*>(myAct + l15 * 64 + t * 16 + mrow) = g;
                }
            }

            // ---- Layer 2: 64x64 ----
            {
                v16h bk0 = *reinterpret_cast<const v16h*>(myAct + l15 * 64 + (hi ? 16 : 0));
                v16h bk1 = *reinterpret_cast<const v16h*>(myAct + l15 * 64 + 32 + (hi ? 16 : 0));
                #pragma unroll 1
                for (int t = 0; t < 4; ++t) {
                    v16h a0 = load_a_tile(sW2, 64, t, 0, lane);
                    v16h a1 = load_a_tile(sW2, 64, t, 1, lane);
                    v8f acc = *reinterpret_cast<const v8f*>(sB2 + t * 16 + mrow);
                    acc = wmma_f16(a0, bk0, acc);
                    acc = wmma_f16(a1, bk1, acc);
                    v8h g;
                    #pragma unroll
                    for (int r = 0; r < 8; ++r) g[r] = (_Float16)gelu_fast(acc[r]);
                    *reinterpret_cast<v8h*>(myAct + l15 * 64 + t * 16 + mrow) = g;
                }
            }

            // ---- Layer 3: 1x64 dot (VALU); lanes c and c+16 split K halves ----
            const int k0 = hi ? 32 : 0;
            float ssum = 0.0f;
            #pragma unroll
            for (int q = 0; q < 4; ++q) {
                v8h av = *reinterpret_cast<const v8h*>(myAct + l15 * 64 + k0 + q * 8);
                v8f wv = *reinterpret_cast<const v8f*>(sW3 + k0 + q * 8);
                #pragma unroll
                for (int r = 0; r < 8; ++r) ssum += wv[r] * (float)av[r];
            }
            ssum += __shfl_xor(ssum, 16, 32);

            if (!hi) {
                float u  = ssum + b3v;
                float nv = sc[c] + dtdx * u;
                nv = fminf(fmaxf(nv, 0.0f), 1.0f);
                sn[c] = nv;
                out[bOff + (size_t)step * NXC + c] = nv;
            }
        }
        __syncthreads();
        p ^= 1;
    }
}

extern "C" void kernel_launch(void* const* d_in, const int* in_sizes, int n_in,
                              void* d_out, int out_size, void* d_ws, size_t ws_size,
                              hipStream_t stream) {
    (void)in_sizes; (void)n_in; (void)d_ws; (void)ws_size; (void)out_size;
    const float* gin = (const float*)d_in[0];
    const float* dtp = (const float*)d_in[1];
    const float* W0  = (const float*)d_in[2];
    const float* b0  = (const float*)d_in[3];
    const float* W1  = (const float*)d_in[4];
    const float* b1  = (const float*)d_in[5];
    const float* W2  = (const float*)d_in[6];
    const float* b2  = (const float*)d_in[7];
    const float* W3  = (const float*)d_in[8];
    const float* b3  = (const float*)d_in[9];
    float* out = (float*)d_out;

    nfv_kernel<<<dim3(NBATCH), dim3(NTHREADS), 0, stream>>>(
        gin, dtp, W0, b0, W1, b1, W2, b2, W3, b3, out);
}
